// AttentionBlock_53558242181799
// MI455X (gfx1250) — matrix operand order repaired and checked by reading, not compiled, not run on hardware
//
#include <hip/hip_runtime.h>

// ---------------------------------------------------------------- constants
#define BATCH   16
#define CIN     512
#define HW      1024
#define OQKV    1536
#define NHEADS  4
#define HD      128
#define NGROUPS 32

typedef __attribute__((ext_vector_type(16))) __bf16        v16bf;
typedef __attribute__((ext_vector_type(8)))  float         v8f;
typedef __attribute__((ext_vector_type(4)))  unsigned int  v4u;

// A/B fragment: 16 bf16 = 32 bytes = two 16B (b128) chunks
union ABFrag {
    v16bf v;
    v4u   q[2];
};

union Pack8 {
    unsigned short u[8];
    v4u            q;
};

__device__ __forceinline__ unsigned short f2bf(float f) {
    unsigned int u = __float_as_uint(f);
    u += 0x7FFFu + ((u >> 16) & 1u);   // round-to-nearest-even
    return (unsigned short)(u >> 16);
}

__device__ __forceinline__ v8f vzero8() {
    v8f z = {0.f, 0.f, 0.f, 0.f, 0.f, 0.f, 0.f, 0.f};
    return z;
}

__device__ __forceinline__ v8f wmma_bf16(const ABFrag& a, const ABFrag& b, v8f c) {
    return __builtin_amdgcn_wmma_f32_16x16x32_bf16(false, a.v, false, b.v,
                                                   (short)0, c, false, false);
}

// CDNA5 async DMA: global -> LDS, 16B per lane, tracked with ASYNCcnt.
__device__ __forceinline__ void async_copy16(unsigned short* lds,
                                             const unsigned short* g) {
    unsigned int l32 = (unsigned int)(unsigned long long)lds;  // LDS byte offset
    asm volatile("global_load_async_to_lds_b128 %0, %1, off"
                 :: "v"(l32), "v"(g) : "memory");
}

__device__ __forceinline__ void wait_async0() {
    asm volatile("s_wait_asynccnt 0x0" ::: "memory");
}

// ------------------------------------------------- kernel 0: fp32 -> bf16
__global__ void convert_bf16_kernel(const float* __restrict__ src,
                                    unsigned short* __restrict__ dst, int n) {
    int i = blockIdx.x * blockDim.x + threadIdx.x;
    if (i < n) dst[i] = f2bf(src[i]);
}

// ------------------------------------------------- kernel 1: GroupNorm -> hT[b][p][c] (bf16)
__global__ void groupnorm_kernel(const float* __restrict__ x,
                                 const float* __restrict__ gamma,
                                 const float* __restrict__ beta,
                                 unsigned short* __restrict__ hT) {
    __shared__ float s1[256];
    __shared__ float s2[256];
    const int g   = blockIdx.x;
    const int b   = blockIdx.y;
    const int tid = threadIdx.x;
    const int CG  = CIN / NGROUPS;                 // 16 channels per group
    const float* xg = x + ((size_t)b * CIN + (size_t)g * CG) * HW;

    float sum = 0.f, sq = 0.f;
    for (int i = tid; i < CG * HW; i += 256) {
        float v = xg[i];
        sum += v; sq += v * v;
    }
    s1[tid] = sum; s2[tid] = sq;
    __syncthreads();
    for (int s = 128; s > 0; s >>= 1) {
        if (tid < s) { s1[tid] += s1[tid + s]; s2[tid] += s2[tid + s]; }
        __syncthreads();
    }
    const float inv_n = 1.f / (float)(CG * HW);
    const float mean  = s1[0] * inv_n;
    const float var   = s2[0] * inv_n - mean * mean;
    const float rstd  = rsqrtf(var + 1e-5f);

    float sc[16], sh[16];
    #pragma unroll
    for (int c = 0; c < 16; c++) {
        float gm = gamma[g * CG + c] * rstd;
        sc[c] = gm;
        sh[c] = beta[g * CG + c] - mean * gm;
    }

    #pragma unroll
    for (int i = 0; i < 4; i++) {
        int p = tid + 256 * i;
        Pack8 lo, hi;
        #pragma unroll
        for (int c = 0; c < 8; c++)  lo.u[c] = f2bf(xg[(size_t)c * HW + p] * sc[c] + sh[c]);
        #pragma unroll
        for (int c = 0; c < 8; c++)  hi.u[c] = f2bf(xg[(size_t)(c + 8) * HW + p] * sc[c + 8] + sh[c + 8]);
        v4u* dst = (v4u*)(hT + ((size_t)b * HW + p) * CIN + (size_t)g * CG);
        dst[0] = lo.q;
        dst[1] = hi.q;
    }
}

// ------------------------------------------------- kernel 2: QKV GEMM (64x64 per wave)
// qkv[o,p] = W[o,:] . h[:,p] + bias[o]
// o < 1024 (q,k)  -> qkT[b][p][o]        (position-major, for QK^T B-frags)
// o >= 1024 (v)   -> vbuf[b][o-1024][p]  (channel-major, for P.V B-frags)
__global__ __launch_bounds__(32)
void qkv_gemm_kernel(const unsigned short* __restrict__ wq,   // [1536][512] bf16
                     const unsigned short* __restrict__ hT,   // [B][1024][512] bf16
                     const float* __restrict__ bias,          // [1536]
                     unsigned short* __restrict__ qkT,        // [B][1024][1024] bf16
                     unsigned short* __restrict__ vbuf) {     // [B][512][1024] bf16
    const int lane = threadIdx.x;
    const int half = lane >> 4;
    const int n    = lane & 15;
    const int klo  = half ? 8 : 0;     // A-frag K origin per half
    const int kb   = half ? 16 : 0;    // unused; B-frag uses klo like A
    const int oBase = blockIdx.x * 64;
    const int pBase = blockIdx.y * 64;
    const int b     = blockIdx.z;

    v8f acc[4][4];
    #pragma unroll
    for (int mt = 0; mt < 4; mt++)
        #pragma unroll
        for (int nt = 0; nt < 4; nt++) acc[mt][nt] = vzero8();

    const unsigned short* hrow = hT + ((size_t)b * HW + pBase + n) * CIN;

    for (int k = 0; k < CIN; k += 32) {
        ABFrag A[4];
        #pragma unroll
        for (int mt = 0; mt < 4; mt++) {
            const unsigned short* wr = wq + (size_t)(oBase + mt * 16 + n) * CIN + k;
            A[mt].q[0] = *(const v4u*)(wr + klo);
            A[mt].q[1] = *(const v4u*)(wr + klo + 16);
        }
        #pragma unroll
        for (int nt = 0; nt < 4; nt++) {
            ABFrag Bt;
            const unsigned short* hr = hrow + (size_t)nt * 16 * CIN + k + klo;
            Bt.q[0] = *(const v4u*)(hr);
            Bt.q[1] = *(const v4u*)(hr + 16);
            #pragma unroll
            for (int mt = 0; mt < 4; mt++)
                acc[mt][nt] = wmma_bf16(A[mt], Bt, acc[mt][nt]);
        }
    }

    #pragma unroll
    for (int mt = 0; mt < 4; mt++) {
        const int oSub = oBase + mt * 16;
        float bv[8];
        #pragma unroll
        for (int r = 0; r < 8; r++) bv[r] = bias[oSub + 8 * half + r];
        #pragma unroll
        for (int nt = 0; nt < 4; nt++) {
            const int p = pBase + nt * 16 + n;
            if (oSub < 1024) {
                Pack8 st;
                #pragma unroll
                for (int r = 0; r < 8; r++) st.u[r] = f2bf(acc[mt][nt][r] + bv[r]);
                *(v4u*)(qkT + ((size_t)b * HW + p) * 1024 + oSub + 8 * half) = st.q;
            } else {
                #pragma unroll
                for (int r = 0; r < 8; r++) {
                    const int vc = oSub - 1024 + 8 * half + r;
                    vbuf[((size_t)b * CIN + vc) * HW + p] = f2bf(acc[mt][nt][r] + bv[r]);
                }
            }
        }
    }
}

// ------------------------------------------------- kernel 3: flash attention
// 4 waves / block = 64 query rows. K/V chunks (32 wide) staged in LDS once per
// block via async-to-LDS DMA, double-buffered so DMA overlaps WMMA + softmax.
#define KROW 136                 // 128 + 8 pad (ushorts) -> 272B rows, bank-safe
#define VROW 40                  // 32 + 8 pad  (ushorts) ->  80B rows, bank-safe
#define PROW 40

__global__ __launch_bounds__(128)
void attn_kernel(const unsigned short* __restrict__ qkT,   // [B][1024][1024]
                 const unsigned short* __restrict__ vbuf,  // [B][512][1024]
                 unsigned short* __restrict__ outT) {      // [B][1024][512]
    __shared__ __align__(16) unsigned short Kbuf[2][32 * KROW];
    __shared__ __align__(16) unsigned short Vbuf[2][128 * VROW];
    __shared__ __align__(16) unsigned short Pbuf[4 * 16 * PROW];

    const int tid   = threadIdx.x;
    const int w     = tid >> 5;          // wave id 0..3
    const int lane  = tid & 31;
    const int half  = lane >> 4;
    const int n     = lane & 15;
    const int klo   = half ? 8 : 0;
    const int kb    = half ? 16 : 0;
    const int iBase = blockIdx.x * 64 + w * 16;
    const int head  = blockIdx.y;
    const int b     = blockIdx.z;
    const float scale = 0.08838834764831845f;   // hd^-0.5

    const unsigned short* kglob = qkT + (size_t)b * HW * 1024 + 512 + head * HD;
    const unsigned short* vhead = vbuf + ((size_t)b * CIN + head * HD) * HW;
    unsigned short* Pw = Pbuf + w * 16 * PROW;

    // Q in A-frag layout: 4 tiles covering c = 0..127 (resident all kernel)
    ABFrag QA[4];
    {
        const unsigned short* qrow = qkT + ((size_t)b * HW + iBase + n) * 1024 + head * HD;
        #pragma unroll
        for (int ct = 0; ct < 4; ct++) {
            QA[ct].q[0] = *(const v4u*)(qrow + ct * 32 + klo);
            QA[ct].q[1] = *(const v4u*)(qrow + ct * 32 + klo + 16);
        }
    }

    v8f O[8];
    #pragma unroll
    for (int c = 0; c < 8; c++) O[c] = vzero8();
    float mrun[8], lrun[8];
    #pragma unroll
    for (int r = 0; r < 8; r++) { mrun[r] = -1e30f; lrun[r] = 0.f; }

    // ---- stage chunk jt into buffer buf: K 32x128, V 128x32 (512 x 16B each)
    auto stage = [&](int buf, int jBase) {
        const unsigned short* krow0 = kglob + (size_t)jBase * 1024;
        const unsigned short* vrow0 = vhead + jBase;
        #pragma unroll
        for (int it = 0; it < 4; it++) {
            int s = tid + 128 * it;
            int row = s >> 4, seg = s & 15;
            async_copy16(&Kbuf[buf][row * KROW + seg * 8],
                         krow0 + (size_t)row * 1024 + seg * 8);
        }
        #pragma unroll
        for (int it = 0; it < 4; it++) {
            int s = tid + 128 * it;
            int c = s >> 2, seg = s & 3;
            async_copy16(&Vbuf[buf][c * VROW + seg * 8],
                         vrow0 + (size_t)c * 1024 + seg * 8);
        }
    };

    stage(0, 0);                         // prologue

    const int NJ = HW / 32;              // 32 chunks
    for (int jt = 0; jt < NJ; jt++) {
        const int cur = jt & 1;
        wait_async0();                   // our chunk-jt copies done
        __syncthreads();                 // everyone's copies done
        if (jt + 1 < NJ) stage(cur ^ 1, (jt + 1) * 32);  // overlap next DMA

        // ---- S = Q K^T from LDS
        v8f S0 = vzero8(), S1 = vzero8();
        const unsigned short* kr0 = &Kbuf[cur][n * KROW];
        const unsigned short* kr1 = &Kbuf[cur][(16 + n) * KROW];
        #pragma unroll
        for (int ct = 0; ct < 4; ct++) {
            ABFrag B0, B1;
            B0.q[0] = *(const v4u*)(kr0 + ct * 32 + klo);
            B0.q[1] = *(const v4u*)(kr0 + ct * 32 + klo + 16);
            B1.q[0] = *(const v4u*)(kr1 + ct * 32 + klo);
            B1.q[1] = *(const v4u*)(kr1 + ct * 32 + klo + 16);
            S0 = wmma_bf16(QA[ct], B0, S0);
            S1 = wmma_bf16(QA[ct], B1, S1);
        }

        // ---- online softmax (rows striped across 16-lane halves)
        float alpha[8];
        #pragma unroll
        for (int r = 0; r < 8; r++) {
            float s0 = S0[r] * scale;
            float s1 = S1[r] * scale;
            float cm = fmaxf(s0, s1);
            cm = fmaxf(cm, __shfl_xor(cm, 1));
            cm = fmaxf(cm, __shfl_xor(cm, 2));
            cm = fmaxf(cm, __shfl_xor(cm, 4));
            cm = fmaxf(cm, __shfl_xor(cm, 8));
            const float mnew = fmaxf(mrun[r], cm);
            const float a    = __expf(mrun[r] - mnew);
            const float e0   = __expf(s0 - mnew);
            const float e1   = __expf(s1 - mnew);
            float rs = e0 + e1;
            rs += __shfl_xor(rs, 1);
            rs += __shfl_xor(rs, 2);
            rs += __shfl_xor(rs, 4);
            rs += __shfl_xor(rs, 8);
            lrun[r] = lrun[r] * a + rs;
            mrun[r] = mnew;
            alpha[r] = a;
            const int mrow = r + 8 * half;
            Pw[mrow * PROW + n]      = f2bf(e0);
            Pw[mrow * PROW + 16 + n] = f2bf(e1);
        }
        #pragma unroll
        for (int c = 0; c < 8; c++)
            #pragma unroll
            for (int r = 0; r < 8; r++)
                O[c][r] *= alpha[r];
        __syncthreads();                 // P visible; K reads done before reuse

        // ---- O += P V from LDS
        ABFrag PA;
        const unsigned short* prow = Pw + n * PROW;
        PA.q[0] = *(const v4u*)(prow + klo);
        PA.q[1] = *(const v4u*)(prow + klo + 16);
        #pragma unroll
        for (int c = 0; c < 8; c++) {
            ABFrag BV;
            const unsigned short* vr = &Vbuf[cur][(c * 16 + n) * VROW];
            BV.q[0] = *(const v4u*)(vr + klo);
            BV.q[1] = *(const v4u*)(vr + klo + 16);
            O[c] = wmma_bf16(PA, BV, O[c]);
        }
    }

    float inv[8];
    #pragma unroll
    for (int r = 0; r < 8; r++) inv[r] = 1.f / lrun[r];
    #pragma unroll
    for (int c = 0; c < 8; c++)
        #pragma unroll
        for (int r = 0; r < 8; r++) {
            const int p = iBase + r + 8 * half;
            outT[((size_t)b * HW + p) * CIN + head * HD + c * 16 + n] =
                f2bf(O[c][r] * inv[r]);
        }
}

// ------------------------------------------------- kernel 4: proj + bias + residual (64x64)
__global__ __launch_bounds__(32)
void proj_kernel(const unsigned short* __restrict__ wp,    // [512][512] bf16
                 const unsigned short* __restrict__ outT,  // [B][1024][512]
                 const float* __restrict__ bias,           // [512]
                 const float* __restrict__ x,              // [B][512][1024]
                 float* __restrict__ y) {                  // [B][512][1024]
    const int lane = threadIdx.x;
    const int half = lane >> 4;
    const int n    = lane & 15;
    const int klo  = half ? 8 : 0;
    const int kb   = half ? 16 : 0;
    const int oBase = blockIdx.x * 64;
    const int pBase = blockIdx.y * 64;
    const int b     = blockIdx.z;

    v8f acc[4][4];
    #pragma unroll
    for (int mt = 0; mt < 4; mt++)
        #pragma unroll
        for (int nt = 0; nt < 4; nt++) acc[mt][nt] = vzero8();

    const unsigned short* orow = outT + ((size_t)b * HW + pBase + n) * CIN;

    for (int k = 0; k < CIN; k += 32) {
        ABFrag A[4];
        #pragma unroll
        for (int mt = 0; mt < 4; mt++) {
            const unsigned short* wr = wp + (size_t)(oBase + mt * 16 + n) * CIN + k;
            A[mt].q[0] = *(const v4u*)(wr + klo);
            A[mt].q[1] = *(const v4u*)(wr + klo + 16);
        }
        #pragma unroll
        for (int nt = 0; nt < 4; nt++) {
            ABFrag Bt;
            const unsigned short* orp = orow + (size_t)nt * 16 * CIN + k + klo;
            Bt.q[0] = *(const v4u*)(orp);
            Bt.q[1] = *(const v4u*)(orp + 16);
            #pragma unroll
            for (int mt = 0; mt < 4; mt++)
                acc[mt][nt] = wmma_bf16(A[mt], Bt, acc[mt][nt]);
        }
    }

    #pragma unroll
    for (int mt = 0; mt < 4; mt++) {
        const int oSub = oBase + mt * 16;
        float bv[8];
        #pragma unroll
        for (int r = 0; r < 8; r++) bv[r] = bias[oSub + 8 * half + r];
        #pragma unroll
        for (int nt = 0; nt < 4; nt++) {
            const int p = pBase + nt * 16 + n;
            #pragma unroll
            for (int r = 0; r < 8; r++) {
                const size_t idx = ((size_t)b * CIN + oSub + 8 * half + r) * HW + p;
                y[idx] = x[idx] + bv[r] + acc[mt][nt][r];
            }
        }
    }
}

// ------------------------------------------------- launch
extern "C" void kernel_launch(void* const* d_in, const int* in_sizes, int n_in,
                              void* d_out, int out_size, void* d_ws, size_t ws_size,
                              hipStream_t stream) {
    const float* x      = (const float*)d_in[0];
    const float* gamma  = (const float*)d_in[1];
    const float* beta   = (const float*)d_in[2];
    const float* w_qkv  = (const float*)d_in[3];
    const float* b_qkv  = (const float*)d_in[4];
    const float* w_proj = (const float*)d_in[5];
    const float* b_proj = (const float*)d_in[6];
    float* out = (float*)d_out;

    char* ws = (char*)d_ws;
    size_t off = 0;
    unsigned short* wqkv_bf  = (unsigned short*)(ws + off); off += (size_t)OQKV * CIN * 2;
    unsigned short* wproj_bf = (unsigned short*)(ws + off); off += (size_t)CIN * CIN * 2;
    unsigned short* hT       = (unsigned short*)(ws + off); off += (size_t)BATCH * HW * CIN * 2;
    unsigned short* qkT      = (unsigned short*)(ws + off); off += (size_t)BATCH * HW * 1024 * 2;
    unsigned short* vbuf     = (unsigned short*)(ws + off); off += (size_t)BATCH * CIN * HW * 2;
    unsigned short* outT     = (unsigned short*)(ws + off); off += (size_t)BATCH * HW * CIN * 2;
    (void)off; (void)ws_size; (void)in_sizes; (void)n_in; (void)out_size;

    const int nWq = OQKV * CIN;
    const int nWp = CIN * CIN;
    convert_bf16_kernel<<<(nWq + 255) / 256, 256, 0, stream>>>(w_qkv, wqkv_bf, nWq);
    convert_bf16_kernel<<<(nWp + 255) / 256, 256, 0, stream>>>(w_proj, wproj_bf, nWp);

    groupnorm_kernel<<<dim3(NGROUPS, BATCH), 256, 0, stream>>>(x, gamma, beta, hT);

    qkv_gemm_kernel<<<dim3(OQKV / 64, HW / 64, BATCH), 32, 0, stream>>>(
        wqkv_bf, hT, b_qkv, qkT, vbuf);

    attn_kernel<<<dim3(HW / 64, NHEADS, BATCH), 128, 0, stream>>>(qkT, vbuf, outT);

    proj_kernel<<<dim3(CIN / 64, HW / 64, BATCH), 32, 0, stream>>>(
        wproj_bf, outT, b_proj, x, out);
}
